// Grok1MoE_850403524958
// MI455X (gfx1250) — compile-verified, matrix operand order checked by reading, not run on hardware
//
#include <hip/hip_runtime.h>

// ---------------------------------------------------------------------------
// Grok1 MoE for MI455X (gfx1250): router + per-expert dual-GEMM + down-proj.
// GEMMs: v_wmma_f32_16x16x32_bf16, double-buffered LDS, async global<->LDS.
// ---------------------------------------------------------------------------

typedef __attribute__((ext_vector_type(16))) __bf16        v16bf;
typedef __attribute__((ext_vector_type(8)))  float         v8f;
typedef __attribute__((ext_vector_type(4)))  unsigned int  u32x4;
typedef __attribute__((ext_vector_type(4)))  int           i32x4;
typedef __attribute__((ext_vector_type(4)))  float         f32x4;

#define D_DIM 2048
#define F_DIM 4096
#define E_NUM 8
#define SOFTCAP 30.0f
#define BK 64

#if __has_builtin(__builtin_amdgcn_global_load_async_to_lds_b128) && \
    __has_builtin(__builtin_amdgcn_global_store_async_from_lds_b128)
#define USE_ASYNC_LDS 1
#else
#define USE_ASYNC_LDS 0
#endif

// Builtin signature (from hipcc diagnostics): first param is `int4 AS(1)*`
// (global), second is the LDS-side pointer. Provide exactly-typed casts.
typedef __attribute__((address_space(1))) i32x4* gvec_ptr;
typedef __attribute__((address_space(3))) i32x4* lvec_ptr;

__device__ __forceinline__ gvec_ptr as_gbl(const void* p) {
  return (gvec_ptr)(unsigned long long)p;
}
__device__ __forceinline__ lvec_ptr as_lds(const void* p) {
  // LDS generic address: low 32 bits are the LDS byte offset.
  return (lvec_ptr)(unsigned int)(unsigned long long)p;
}

__device__ __forceinline__ void wait_async() {
#if USE_ASYNC_LDS
#if __has_builtin(__builtin_amdgcn_s_wait_asynccnt)
  __builtin_amdgcn_s_wait_asynccnt(0);
#else
  asm volatile("s_wait_asynccnt 0x0" ::: "memory");
#endif
#endif
}

__device__ __forceinline__ void copy16_g2l(unsigned short* lds_dst,
                                           const unsigned short* gbl_src) {
#if USE_ASYNC_LDS
  __builtin_amdgcn_global_load_async_to_lds_b128(as_gbl(gbl_src), as_lds(lds_dst), 0, 0);
#else
  *reinterpret_cast<u32x4*>(lds_dst) = *reinterpret_cast<const u32x4*>(gbl_src);
#endif
}

__device__ __forceinline__ void copy16_l2g(unsigned short* gbl_dst,
                                           const unsigned short* lds_src) {
#if USE_ASYNC_LDS
  __builtin_amdgcn_global_store_async_from_lds_b128(as_gbl(gbl_dst), as_lds(lds_src), 0, 0);
#else
  *reinterpret_cast<u32x4*>(gbl_dst) = *reinterpret_cast<const u32x4*>(lds_src);
#endif
}

__device__ __forceinline__ unsigned short f2bf(float f) {
  unsigned int u = __builtin_bit_cast(unsigned int, f);
  u += 0x7fffu + ((u >> 16) & 1u);          // round-to-nearest-even
  return (unsigned short)(u >> 16);
}

__device__ __forceinline__ float gelu_tanh(float x) {
  float x3 = x * x * x;
  return 0.5f * x * (1.0f + tanhf(0.7978845608028654f * (x + 0.044715f * x3)));
}

// WMMA fragment: 16 bf16 per lane = 8 VGPRs, filled by two 16B LDS loads.
struct Frag { union { u32x4 q[2]; v16bf v; }; };

__device__ __forceinline__ void load_frag(Frag& fr, const unsigned short* p) {
  fr.q[0] = *reinterpret_cast<const u32x4*>(p);
  fr.q[1] = *reinterpret_cast<const u32x4*>(p + 16);
}

// ---------------------------------------------------------------------------
// Kernel 0: fp32 -> bf16 convert of hidden states (read once, reuse 8x).
// ---------------------------------------------------------------------------
__global__ void __launch_bounds__(256)
convert_bf16_kernel(const float* __restrict__ x, unsigned short* __restrict__ xb, long n) {
  long i = ((long)blockIdx.x * blockDim.x + threadIdx.x) * 4;
  if (i + 3 < n) {
    f32x4 v = *reinterpret_cast<const f32x4*>(x + i);
    unsigned int lo = (unsigned int)f2bf(v[0]) | ((unsigned int)f2bf(v[1]) << 16);
    unsigned int hi = (unsigned int)f2bf(v[2]) | ((unsigned int)f2bf(v[3]) << 16);
    *reinterpret_cast<unsigned int*>(xb + i)     = lo;
    *reinterpret_cast<unsigned int*>(xb + i + 2) = hi;
  }
}

// ---------------------------------------------------------------------------
// Kernel 1: router. One wave32 per token.
// ---------------------------------------------------------------------------
__global__ void __launch_bounds__(256)
router_kernel(const float* __restrict__ x, const float* __restrict__ gw,
              const int* __restrict__ topk_p, float* __restrict__ combine, int T) {
  const int wave = threadIdx.x >> 5;
  const int lane = threadIdx.x & 31;
  const int t = blockIdx.x * 8 + wave;
  if (t >= T) return;

  float acc[E_NUM];
#pragma unroll
  for (int e = 0; e < E_NUM; ++e) acc[e] = 0.0f;

  const float* xrow = x + (size_t)t * D_DIM;
  for (int d = lane; d < D_DIM; d += 32) {
    float xv = xrow[d];
    const float* g = gw + (size_t)d * E_NUM;
#pragma unroll
    for (int e = 0; e < E_NUM; ++e) acc[e] += xv * g[e];
  }
#pragma unroll
  for (int e = 0; e < E_NUM; ++e) {
#pragma unroll
    for (int off = 16; off > 0; off >>= 1)
      acc[e] += __shfl_down(acc[e], off, 32);
  }

  if (lane == 0) {
    float l[E_NUM], p[E_NUM];
    float mx = -1e30f;
#pragma unroll
    for (int e = 0; e < E_NUM; ++e) {
      l[e] = tanhf(acc[e] * (1.0f / SOFTCAP)) * SOFTCAP;
      mx = fmaxf(mx, l[e]);
    }
    float sum = 0.0f;
#pragma unroll
    for (int e = 0; e < E_NUM; ++e) { p[e] = __expf(l[e] - mx); sum += p[e]; }
    float inv = 1.0f / sum;
#pragma unroll
    for (int e = 0; e < E_NUM; ++e) p[e] *= inv;

    int k = *topk_p;
    bool sel[E_NUM];
#pragma unroll
    for (int e = 0; e < E_NUM; ++e) sel[e] = false;
    for (int s = 0; s < k; ++s) {
      int bi = 0; float bv = -1.0f;
#pragma unroll
      for (int e = 0; e < E_NUM; ++e)
        if (!sel[e] && p[e] > bv) { bv = p[e]; bi = e; }
      sel[bi] = true;
    }
    float* crow = combine + (size_t)t * E_NUM;
#pragma unroll
    for (int e = 0; e < E_NUM; ++e) crow[e] = sel[e] ? p[e] : 0.0f;
  }
}

// ---------------------------------------------------------------------------
// Kernel 2 (per expert): H = gelu(X @ Wg) * (X @ Wu), bf16 out.
// Block 128(M) x 64(N), BK=64, double-buffered LDS, 8 waves as 4(M) x 2(N),
// wave tile 32x32 per output. A tile moved with async global->LDS; H tile
// staged in LDS and written with async LDS->global b128 stores.
// ---------------------------------------------------------------------------
__global__ void __launch_bounds__(256)
moe_pass1(const unsigned short* __restrict__ xb,
          const float* __restrict__ wg, const float* __restrict__ wu,
          unsigned short* __restrict__ H) {
  __shared__ union {
    struct {
      unsigned short A [2][128 * BK];   // [m][k]  32 KB
      unsigned short Bg[2][64 * BK];    // [n][k]  16 KB
      unsigned short Bu[2][64 * BK];    // [n][k]  16 KB
    } t;
    unsigned short Hs[128 * 64];        // epilogue staging, 16 KB
  } sm;

  const int tid  = threadIdx.x;
  const int lane = tid & 31;
  const int wave = tid >> 5;
  const int m0 = blockIdx.y * 128;
  const int f0 = blockIdx.x * 64;
  const int m_off = (wave >> 1) * 32;
  const int n_off = (wave & 1) * 32;
  const int r  = lane & 15;
  const int kh = (lane >> 4) << 3;      // K-half select: 0 or 8

  v8f accg[2][2], accu[2][2];
#pragma unroll
  for (int i = 0; i < 2; ++i)
#pragma unroll
    for (int j = 0; j < 2; ++j) { accg[i][j] = {}; accu[i][j] = {}; }

  auto load_A = [&](int k0, int buf) {
#pragma unroll
    for (int i = 0; i < 4; ++i) {       // 128 rows x 8 chunks(16B) = 1024
      const int idx = tid + 256 * i;
      const int row = idx >> 3;
      const int ce  = (idx & 7) * 8;
      copy16_g2l(&sm.t.A[buf][row * BK + ce],
                 xb + (size_t)(m0 + row) * D_DIM + k0 + ce);
    }
  };
  auto load_B = [&](int k0, int buf) {
    const int fr4 = (tid & 15) * 4;     // f column base (0..60)
    const int d0  = tid >> 4;           // k row (0..15), + 16*dd
#pragma unroll
    for (int dd = 0; dd < 4; ++dd) {
      const int d = d0 + dd * 16;
      const size_t go = (size_t)(k0 + d) * F_DIM + f0 + fr4;
      f32x4 vg = *reinterpret_cast<const f32x4*>(wg + go);
      f32x4 vu = *reinterpret_cast<const f32x4*>(wu + go);
#pragma unroll
      for (int jj = 0; jj < 4; ++jj) {
        sm.t.Bg[buf][(fr4 + jj) * BK + d] = f2bf(vg[jj]);
        sm.t.Bu[buf][(fr4 + jj) * BK + d] = f2bf(vu[jj]);
      }
    }
  };

  load_A(0, 0); load_B(0, 0);
  wait_async();
  __syncthreads();

  int buf = 0;
  for (int k0 = 0; k0 < D_DIM; k0 += BK, buf ^= 1) {
    if (k0 + BK < D_DIM) { load_A(k0 + BK, buf ^ 1); load_B(k0 + BK, buf ^ 1); }
#pragma unroll
    for (int kk = 0; kk < BK; kk += 32) {
      Frag a[2], bg[2], bu[2];
#pragma unroll
      for (int i = 0; i < 2; ++i)
        load_frag(a[i], &sm.t.A[buf][(m_off + i * 16 + r) * BK + kk + kh]);
#pragma unroll
      for (int j = 0; j < 2; ++j) {
        load_frag(bg[j], &sm.t.Bg[buf][(n_off + j * 16 + r) * BK + kk + kh]);
        load_frag(bu[j], &sm.t.Bu[buf][(n_off + j * 16 + r) * BK + kk + kh]);
      }
#pragma unroll
      for (int i = 0; i < 2; ++i)
#pragma unroll
        for (int j = 0; j < 2; ++j) {
          accg[i][j] = __builtin_amdgcn_wmma_f32_16x16x32_bf16(
              false, a[i].v, false, bg[j].v, (short)0, accg[i][j], false, false);
          accu[i][j] = __builtin_amdgcn_wmma_f32_16x16x32_bf16(
              false, a[i].v, false, bu[j].v, (short)0, accu[i][j], false, false);
        }
    }
    wait_async();
    __syncthreads();
  }

  // Epilogue: H = gelu(hg)*hu -> stage tile in LDS -> 16B (async) stores.
  {
    const int rb = m_off + ((lane < 16) ? 0 : 8);
    const int cb = n_off + (lane & 15);
#pragma unroll
    for (int i = 0; i < 2; ++i)
#pragma unroll
      for (int j = 0; j < 2; ++j)
#pragma unroll
        for (int rr = 0; rr < 8; ++rr) {
          const float h = gelu_tanh(accg[i][j][rr]) * accu[i][j][rr];
          sm.Hs[(rb + i * 16 + rr) * 64 + cb + j * 16] = f2bf(h);
        }
  }
  __syncthreads();
#pragma unroll
  for (int i = 0; i < 4; ++i) {         // 128 rows x 8 chunks(16B) = 1024
    const int idx = tid + 256 * i;
    const int row = idx >> 3;
    const int ce  = (idx & 7) * 8;
    copy16_l2g(H + (size_t)(m0 + row) * F_DIM + f0 + ce,
               &sm.Hs[row * 64 + ce]);
  }
  // S_ENDPGM performs an implicit wait-idle, draining outstanding async stores.
}

// ---------------------------------------------------------------------------
// Kernel 3 (per expert): out (+)= diag(combine[:,e]) * (H @ Wd).
// Block 128(M) x 128(N), BK=64, double-buffered; 8 waves as 4(M) x 2(N);
// wave tile 32x64.
// ---------------------------------------------------------------------------
__global__ void __launch_bounds__(256)
moe_pass2(const unsigned short* __restrict__ H, const float* __restrict__ wd,
          const float* __restrict__ combine, int expert, int beta,
          float* __restrict__ out) {
  __shared__ unsigned short sA[2][128 * BK];  // [m][k] 32 KB
  __shared__ unsigned short sB[2][128 * BK];  // [n][k] 32 KB

  const int tid  = threadIdx.x;
  const int lane = tid & 31;
  const int wave = tid >> 5;
  const int m0 = blockIdx.y * 128;
  const int n0 = blockIdx.x * 128;
  const int m_off = (wave >> 1) * 32;
  const int n_off = (wave & 1) * 64;
  const int r  = lane & 15;
  const int kh = (lane >> 4) << 3;

  v8f acc[2][4];
#pragma unroll
  for (int i = 0; i < 2; ++i)
#pragma unroll
    for (int j = 0; j < 4; ++j) acc[i][j] = {};

  auto load_A = [&](int k0, int buf) {
#pragma unroll
    for (int i = 0; i < 4; ++i) {
      const int idx = tid + 256 * i;
      const int row = idx >> 3;
      const int ce  = (idx & 7) * 8;
      copy16_g2l(&sA[buf][row * BK + ce],
                 H + (size_t)(m0 + row) * F_DIM + k0 + ce);
    }
  };
  auto load_B = [&](int k0, int buf) {
    const int c4 = (tid & 31) * 4;      // d column base (0..124)
    const int d0 = tid >> 5;            // k row (0..7), + 8*dd
#pragma unroll
    for (int dd = 0; dd < 8; ++dd) {
      const int d = d0 + dd * 8;
      const size_t go = (size_t)(k0 + d) * D_DIM + n0 + c4;
      f32x4 v = *reinterpret_cast<const f32x4*>(wd + go);
#pragma unroll
      for (int jj = 0; jj < 4; ++jj)
        sB[buf][(c4 + jj) * BK + d] = f2bf(v[jj]);
    }
  };

  load_A(0, 0); load_B(0, 0);
  wait_async();
  __syncthreads();

  int buf = 0;
  for (int k0 = 0; k0 < F_DIM; k0 += BK, buf ^= 1) {
    if (k0 + BK < F_DIM) { load_A(k0 + BK, buf ^ 1); load_B(k0 + BK, buf ^ 1); }
#pragma unroll
    for (int kk = 0; kk < BK; kk += 32) {
      Frag a[2], b[4];
#pragma unroll
      for (int i = 0; i < 2; ++i)
        load_frag(a[i], &sA[buf][(m_off + i * 16 + r) * BK + kk + kh]);
#pragma unroll
      for (int j = 0; j < 4; ++j)
        load_frag(b[j], &sB[buf][(n_off + j * 16 + r) * BK + kk + kh]);
#pragma unroll
      for (int i = 0; i < 2; ++i)
#pragma unroll
        for (int j = 0; j < 4; ++j)
          acc[i][j] = __builtin_amdgcn_wmma_f32_16x16x32_bf16(
              false, a[i].v, false, b[j].v, (short)0, acc[i][j], false, false);
    }
    wait_async();
    __syncthreads();
  }

  const int row_base = m0 + m_off + ((lane < 16) ? 0 : 8);
  const int col_base = n0 + n_off + (lane & 15);
#pragma unroll
  for (int i = 0; i < 2; ++i)
#pragma unroll
    for (int rr = 0; rr < 8; ++rr) {
      const int row = row_base + i * 16 + rr;
      const float c = combine[(size_t)row * E_NUM + expert];
#pragma unroll
      for (int j = 0; j < 4; ++j) {
        const int col = col_base + j * 16;
        const size_t o = (size_t)row * D_DIM + col;
        const float v = c * acc[i][j][rr];
        out[o] = beta ? (out[o] + v) : v;
      }
    }
}

// ---------------------------------------------------------------------------
// Host launcher
// ---------------------------------------------------------------------------
extern "C" void kernel_launch(void* const* d_in, const int* in_sizes, int n_in,
                              void* d_out, int out_size, void* d_ws, size_t ws_size,
                              hipStream_t stream) {
  const float* x      = (const float*)d_in[0];   // [T, D] fp32
  const float* gate_w = (const float*)d_in[1];   // [D, E] fp32
  const float* w_gate = (const float*)d_in[2];   // [E, D, F] fp32
  const float* w_up   = (const float*)d_in[3];   // [E, D, F] fp32
  const float* w_down = (const float*)d_in[4];   // [E, F, D] fp32
  const int*   top_k  = (const int*)d_in[5];     // scalar
  float* out = (float*)d_out;                    // [T, D] fp32

  const int T = in_sizes[0] / D_DIM;             // 8192

  // Workspace layout: x_bf16 [T,D] | combine [T,E] fp32 | H [T,F] bf16
  char* ws = (char*)d_ws;
  unsigned short* xb = (unsigned short*)ws;
  size_t off = (size_t)T * D_DIM * sizeof(unsigned short);
  float* combine = (float*)(ws + off);
  off += (size_t)T * E_NUM * sizeof(float);
  unsigned short* Hbuf = (unsigned short*)(ws + off);

  const long nX = (long)T * D_DIM;
  convert_bf16_kernel<<<(unsigned)((nX + 1023) / 1024), 256, 0, stream>>>(x, xb, nX);
  router_kernel<<<(unsigned)((T + 7) / 8), 256, 0, stream>>>(x, gate_w, top_k, combine, T);

  const dim3 g1(F_DIM / 64, T / 128);
  const dim3 g2(D_DIM / 128, T / 128);
  for (int e = 0; e < E_NUM; ++e) {
    moe_pass1<<<g1, 256, 0, stream>>>(xb,
        w_gate + (size_t)e * D_DIM * F_DIM,
        w_up   + (size_t)e * D_DIM * F_DIM, Hbuf);
    moe_pass2<<<g2, 256, 0, stream>>>(Hbuf,
        w_down + (size_t)e * F_DIM * D_DIM, combine, e, e > 0 ? 1 : 0, out);
  }
}
